// LoopClosureUNet_87565793231058
// MI455X (gfx1250) — matrix operand order repaired and checked by reading, not compiled, not run on hardware
//
#include <hip/hip_runtime.h>
#include <hip/hip_bf16.h>

typedef __attribute__((ext_vector_type(16))) _Float16 v16h;
typedef __attribute__((ext_vector_type(8)))  _Float16 v8h;
typedef __attribute__((ext_vector_type(8)))  float    v8f;

#define HD 128      // hidden width (all GCN outputs)
#define NG 8        // num graphs
#define CDIV(a,b) ((int)(((a)+(b)-1)/(b)))

// ---------------------------------------------------------------------------
// f16 WMMA GEMM: C(f32, MxN) = A(f16, MxK row-major) @ Bt^T
// where Bt is the N x K row-major TRANSPOSE of B. All of M, N, K must be
// multiples of 64 (true for every GEMM in this pipeline) -> no bounds guards,
// fully vectorized b128 staging, and every WMMA fragment is 2x ds_load_b128.
// 128 threads = 4 waves; block tile 64x64; wave tile 32x32 (2x2 fragments,
// each A/B fragment feeds two v_wmma_f32_16x16x32_f16).
// ---------------------------------------------------------------------------
#define TM 64
#define TN 64
#define TK 64
#define LDT (TK + 8)   // LDS row stride in halves (multiple of 8 -> 16B align)

// A-operand fragment: elems 0..7 = K kb..kb+7, elems 8..15 = K kb+16..kb+23
__device__ __forceinline__ v16h frag_gap(const _Float16* p) {
  v8h lo = *(const v8h*)p;
  v8h hi = *(const v8h*)(p + 16);
  return __builtin_shufflevector(lo, hi, 0,1,2,3,4,5,6,7,8,9,10,11,12,13,14,15);
}
// Unused: 16 contiguous K values per half-wave (B also uses frag_gap)
__device__ __forceinline__ v16h frag_cont(const _Float16* p) {
  v8h lo = *(const v8h*)p;
  v8h hi = *(const v8h*)(p + 8);
  return __builtin_shufflevector(lo, hi, 0,1,2,3,4,5,6,7,8,9,10,11,12,13,14,15);
}

__global__ __launch_bounds__(128) void k_wmma_gemm(
    const _Float16* __restrict__ A, const _Float16* __restrict__ Bt,
    float* __restrict__ C, int M, int N, int K) {
  __shared__ _Float16 As[TM][LDT];
  __shared__ _Float16 Bs[TN][LDT];

  const int tid  = threadIdx.x;
  const int lane = tid & 31;
  const int wave = tid >> 5;
  const int wm = (wave >> 1) * 32;
  const int wn = (wave & 1)  * 32;
  const long tileM = (long)blockIdx.y * TM;
  const long tileN = (long)blockIdx.x * TN;

  // staging assignment: 2 threads per row, 32 halves (64B) each
  const int sr = tid >> 1;
  const int sc = (tid & 1) * 32;
  const _Float16* ga = A  + (tileM + sr) * (long)K + sc;
  const _Float16* gb = Bt + (tileN + sr) * (long)K + sc;

  v8f acc00 = {}, acc01 = {}, acc10 = {}, acc11 = {};

  const int fr  = lane & 15;
  const int kba = (lane >> 4) * 8;     // A: half-wave K interleave of 8
  const int kbb = (lane >> 4) * 16;    // unused: B is read with kba, like A

  for (int k0 = 0; k0 < K; k0 += TK) {
#pragma unroll
    for (int h = 0; h < 4; ++h) {
      *(v8h*)&As[sr][sc + h * 8] = *(const v8h*)(ga + k0 + h * 8);
      *(v8h*)&Bs[sr][sc + h * 8] = *(const v8h*)(gb + k0 + h * 8);
    }
    if (k0 + TK < K) {
      __builtin_prefetch(ga + k0 + TK, 0, 0);
      __builtin_prefetch(gb + k0 + TK, 0, 0);
    }
    __syncthreads();

#pragma unroll
    for (int kk = 0; kk < TK; kk += 32) {
      v16h a0 = frag_gap (&As[wm + fr     ][kk + kba]);
      v16h a1 = frag_gap (&As[wm + 16 + fr][kk + kba]);
      v16h b0 = frag_gap (&Bs[wn + fr     ][kk + kba]);
      v16h b1 = frag_gap (&Bs[wn + 16 + fr][kk + kba]);
      acc00 = __builtin_amdgcn_wmma_f32_16x16x32_f16(false, a0, false, b0, (short)0, acc00, false, false);
      acc01 = __builtin_amdgcn_wmma_f32_16x16x32_f16(false, a0, false, b1, (short)0, acc01, false, false);
      acc10 = __builtin_amdgcn_wmma_f32_16x16x32_f16(false, a1, false, b0, (short)0, acc10, false, false);
      acc11 = __builtin_amdgcn_wmma_f32_16x16x32_f16(false, a1, false, b1, (short)0, acc11, false, false);
    }
    __syncthreads();
  }

  // C layout: VGPR e -> row = e + 8*(lane>>4), col = lane&15
  const int cr = (lane >> 4) * 8;
  const int cc = lane & 15;
#pragma unroll
  for (int e = 0; e < 8; ++e) {
    const long r0 = tileM + wm + cr + e;
    const long r1 = r0 + 16;
    const long c0 = tileN + wn + cc;
    C[r0 * N + c0]      = acc00[e];
    C[r0 * N + c0 + 16] = acc01[e];
    C[r1 * N + c0]      = acc10[e];
    C[r1 * N + c0 + 16] = acc11[e];
  }
}

// ---------------------------------------------------------------------------
// Bandwidth-bound glue kernels
// ---------------------------------------------------------------------------
__global__ void k_fill_f32(float* p, long n, float v) {
  long t = (long)blockIdx.x * blockDim.x + threadIdx.x;
  if (t < n) p[t] = v;
}

__global__ void k_build_adj(const int* __restrict__ ei, float* A, int n, int ne) {
  int t = blockIdx.x * blockDim.x + threadIdx.x;
  if (t < ne) {
    int src = ei[t];          // row 0 of edge_index
    int dst = ei[ne + t];     // row 1
    atomicAdd(&A[(long)dst * n + src], 1.0f);
  }
}

__global__ void k_f32_to_f16(const float* __restrict__ x, _Float16* __restrict__ y, long n) {
  long t = (long)blockIdx.x * blockDim.x + threadIdx.x;
  if (t < n) y[t] = (_Float16)x[t];
}

// Wt[j*fin + k] = W[k*HD + j]   (emit B pre-transposed for the GEMM)
__global__ void k_transpose_w(const float* __restrict__ W, _Float16* __restrict__ Wt, int fin) {
  int t = blockIdx.x * blockDim.x + threadIdx.x;
  if (t < fin * HD) {
    int j = t / fin, k = t - j * fin;
    Wt[t] = (_Float16)W[k * HD + j];
  }
}

__global__ void k_add_eye_f16(const float* __restrict__ A, _Float16* __restrict__ S, int n) {
  long t = (long)blockIdx.x * blockDim.x + threadIdx.x;
  long total = (long)n * n;
  if (t < total) {
    int r = (int)(t / n), c = (int)(t - (long)r * n);
    S[t] = (_Float16)(A[t] + (r == c ? 1.0f : 0.0f));
  }
}

__global__ void k_zero_diag(float* A, int n) {
  int t = blockIdx.x * blockDim.x + threadIdx.x;
  if (t < n) A[(long)t * n + t] = 0.0f;
}

__global__ void k_rowsum_rsqrt(const float* __restrict__ A, float* d, int n) {
  __shared__ float red[128];
  const int row = blockIdx.x;
  float s = 0.0f;
  for (int j = threadIdx.x; j < n; j += 128) s += A[(long)row * n + j];
  red[threadIdx.x] = s;
  __syncthreads();
  for (int off = 64; off > 0; off >>= 1) {
    if (threadIdx.x < off) red[threadIdx.x] += red[threadIdx.x + off];
    __syncthreads();
  }
  if (threadIdx.x == 0) d[row] = rsqrtf(red[0] + 2.0f);  // A_hat = A + 2I
}

// Z = d (.) XW ; also emit Z^T as f16 (B operand of the A@Z GEMM)
__global__ void k_scale_rows(const float* __restrict__ XW, const float* __restrict__ d,
                             float* __restrict__ Z, _Float16* __restrict__ Zt, int n) {
  long t = (long)blockIdx.x * blockDim.x + threadIdx.x;
  if (t < (long)n * HD) {
    int r = (int)(t >> 7), c = (int)(t & 127);
    float z = d[r] * XW[t];
    Z[t] = z;
    Zt[(long)c * n + r] = (_Float16)z;
  }
}

__global__ void k_gcn_epilogue(const float* __restrict__ AZ, const float* __restrict__ Z,
                               const float* __restrict__ d, const float* __restrict__ bias,
                               float* __restrict__ Y, long nh, int relu) {
  long t = (long)blockIdx.x * blockDim.x + threadIdx.x;
  if (t < nh) {
    int r = (int)(t >> 7), c = (int)(t & 127);
    float y = d[r] * (AZ[t] + 2.0f * Z[t]) + bias[c];
    Y[t] = relu ? fmaxf(y, 0.0f) : y;
  }
}

__global__ void k_invnorm(const float* __restrict__ w, float* invn) {
  __shared__ float red[HD];
  int j = threadIdx.x;
  float v = w[j];
  red[j] = v * v;
  __syncthreads();
  for (int s = 64; s > 0; s >>= 1) {
    if (j < s) red[j] += red[j + s];
    __syncthreads();
  }
  if (j == 0) invn[0] = rsqrtf(red[0]);
}

__global__ void k_scores(const float* __restrict__ h, const float* __restrict__ w,
                         const float* __restrict__ invn, float* sc, int n) {
  int t = blockIdx.x * blockDim.x + threadIdx.x;
  if (t < n) {
    float s = 0.0f;
    const float* row = h + (long)t * HD;
    for (int k = 0; k < HD; ++k) s += row[k] * w[k];
    sc[t] = tanhf(s * invn[0]);
  }
}

// In-LDS bitonic sort (descending), n is a power of two <= 4096.
__global__ __launch_bounds__(1024) void k_bitonic(const float* __restrict__ score, int n, int* perm) {
  __shared__ float ks[4096];
  __shared__ int   vs[4096];
  const int tid = threadIdx.x;
  for (int i = tid; i < n; i += 1024) { ks[i] = score[i]; vs[i] = i; }
  __syncthreads();
  for (int size = 2; size <= n; size <<= 1) {
    for (int stride = size >> 1; stride > 0; stride >>= 1) {
      for (int i = tid; i < n; i += 1024) {
        int j = i ^ stride;
        if (j > i) {
          float a = ks[i], b = ks[j];
          bool desc = ((i & size) == 0);
          if (desc ? (a < b) : (a > b)) {
            ks[i] = b; ks[j] = a;
            int tv = vs[i]; vs[i] = vs[j]; vs[j] = tv;
          }
        }
      }
      __syncthreads();
    }
  }
  for (int i = tid; i < n; i += 1024) perm[i] = vs[i];
}

__global__ void k_gather_pool(const float* __restrict__ h, const float* __restrict__ sc,
                              const int* __restrict__ perm, float* __restrict__ o, int k) {
  long t = (long)blockIdx.x * blockDim.x + threadIdx.x;
  if (t < (long)k * HD) {
    int r = (int)(t >> 7), c = (int)(t & 127);
    int p = perm[r];
    o[t] = h[(long)p * HD + c] * sc[p];
  }
}

__global__ void k_gather_adj(const float* __restrict__ Asq, const int* __restrict__ perm,
                             float* __restrict__ Ao, int n, int k) {
  long t = (long)blockIdx.x * blockDim.x + threadIdx.x;
  if (t < (long)k * k) {
    int r = (int)(t / k), c = (int)(t - (long)r * k);
    Ao[t] = Asq[(long)perm[r] * n + perm[c]];
  }
}

__global__ void k_scatter_add(float* __restrict__ dst, const float* __restrict__ src,
                              const int* __restrict__ perm, int k) {
  long t = (long)blockIdx.x * blockDim.x + threadIdx.x;
  if (t < (long)k * HD) {
    int r = (int)(t >> 7), c = (int)(t & 127);
    dst[(long)perm[r] * HD + c] += src[t];
  }
}

__global__ void k_segsum(const float* __restrict__ h, const int* __restrict__ batch,
                         float* g, int n) {
  long t = (long)blockIdx.x * blockDim.x + threadIdx.x;
  if (t < (long)n * HD) {
    int r = (int)(t >> 7), c = (int)(t & 127);
    atomicAdd(&g[batch[r] * HD + c], h[t]);
  }
}

// 8x128 head: 3x (BN -> linear -> tanh), final BN, 128x32 output linear.
__global__ __launch_bounds__(128) void k_head(const float* __restrict__ g_in,
                                              const float* __restrict__ lw,
                                              const float* __restrict__ lb,
                                              const float* __restrict__ bng,
                                              const float* __restrict__ bnb,
                                              const float* __restrict__ ow,
                                              const float* __restrict__ ob,
                                              float* __restrict__ out) {
  __shared__ float gb[NG][HD];
  __shared__ float tb[NG][HD];
  const int j = threadIdx.x;
  const float inv = rsqrtf(1.0f + 1e-5f);
  for (int b = 0; b < NG; ++b) gb[b][j] = g_in[b * HD + j];
  __syncthreads();
  for (int L = 0; L < 3; ++L) {
    float gam = bng[L * HD + j] * inv;
    float bet = bnb[L * HD + j];
    for (int b = 0; b < NG; ++b) gb[b][j] = gb[b][j] * gam + bet;
    __syncthreads();
    const float* W = lw + L * HD * HD;
    float bias = lb[L * HD + j];
    for (int b = 0; b < NG; ++b) {
      float s = bias;
      for (int k = 0; k < HD; ++k) s += gb[b][k] * W[k * HD + j];
      tb[b][j] = tanhf(s);
    }
    __syncthreads();
    for (int b = 0; b < NG; ++b) gb[b][j] = tb[b][j];
    __syncthreads();
  }
  float gam = bng[3 * HD + j] * inv;
  float bet = bnb[3 * HD + j];
  for (int b = 0; b < NG; ++b) gb[b][j] = gb[b][j] * gam + bet;
  __syncthreads();
  if (j < 32) {
    for (int b = 0; b < NG; ++b) {
      float s = ob[j];
      for (int k = 0; k < HD; ++k) s += gb[b][k] * ow[k * 32 + j];
      out[b * 32 + j] = s;
    }
  }
}

// ---------------------------------------------------------------------------
// Host orchestration
// ---------------------------------------------------------------------------
extern "C" void kernel_launch(void* const* d_in, const int* in_sizes, int n_in,
                              void* d_out, int out_size, void* d_ws, size_t ws_size,
                              hipStream_t stream) {
  (void)n_in; (void)out_size; (void)ws_size;
  const float* x       = (const float*)d_in[0];
  const int*   ei      = (const int*)d_in[1];
  const int*   batch   = (const int*)d_in[2];
  const float* conv0_w = (const float*)d_in[3];
  const float* conv0_b = (const float*)d_in[4];
  const float* down_w  = (const float*)d_in[5];
  const float* down_b  = (const float*)d_in[6];
  const float* pool_w  = (const float*)d_in[7];
  const float* up_w    = (const float*)d_in[8];
  const float* up_b    = (const float*)d_in[9];
  const float* lins_w  = (const float*)d_in[10];
  const float* lins_b  = (const float*)d_in[11];
  const float* bn_g    = (const float*)d_in[12];
  const float* bn_b    = (const float*)d_in[13];
  const float* out_w   = (const float*)d_in[14];
  const float* out_b   = (const float*)d_in[15];
  float* out = (float*)d_out;
  const int ne = in_sizes[1] / 2;

  // workspace carve-out (~203 MB)
  char* base = (char*)d_ws;
  size_t off = 0;
  auto alloc = [&](size_t bytes) -> char* {
    char* p = base + off;
    off = (off + bytes + 255) & ~(size_t)255;
    return p;
  };
  float* A0  = (float*)alloc(4096UL * 4096 * 4);
  float* A1  = (float*)alloc(2048UL * 2048 * 4);
  float* A2  = (float*)alloc(1024UL * 1024 * 4);
  float* A3  = (float*)alloc(512UL * 512 * 4);
  float* ASQ = (float*)alloc(4096UL * 4096 * 4);
  _Float16* AH = (_Float16*)alloc(4096UL * 4096 * 2);
  float* XS0 = (float*)alloc(4096UL * HD * 4);
  float* XS1 = (float*)alloc(2048UL * HD * 4);
  float* XS2 = (float*)alloc(1024UL * HD * 4);
  float* HC  = (float*)alloc(4096UL * HD * 4);
  float* HT  = (float*)alloc(4096UL * HD * 4);
  float* XW  = (float*)alloc(4096UL * HD * 4);
  float* Z   = (float*)alloc(4096UL * HD * 4);
  _Float16* ZHT = (_Float16*)alloc(4096UL * HD * 2);   // Z^T (HD x n), f16
  _Float16* XH  = (_Float16*)alloc(4096UL * HD * 2);   // X (n x fin), f16
  _Float16* WHT = (_Float16*)alloc((size_t)HD * HD * 2); // W^T (HD x fin), f16
  float* D   = (float*)alloc(4096 * 4);
  float* SC  = (float*)alloc(4096 * 4);
  float* INV = (float*)alloc(256);
  int* P0 = (int*)alloc(4096 * 4);
  int* P1 = (int*)alloc(2048 * 4);
  int* P2 = (int*)alloc(1024 * 4);
  float* G = (float*)alloc((size_t)NG * HD * 4);

  // C(MxN) = A(MxK) @ Bt^T ; Bt is N x K row-major
  auto gemm = [&](const _Float16* Am, const _Float16* Bt, float* Cm, int M, int N, int K) {
    dim3 grid(CDIV(N, TN), CDIV(M, TM));
    k_wmma_gemm<<<grid, dim3(128), 0, stream>>>(Am, Bt, Cm, M, N, K);
  };

  // GCN(improved): Y = d (.) (A @ (d (.) (X W)) + 2 (d (.) (X W))) + b
  // NOTE: adjacency matrices here are symmetric; A itself is the A-operand and Z^T
  // is the transposed B-operand in the message-passing GEMM.
  auto gcn = [&](const float* A, const float* X, const float* W, const float* bias,
                 int n, int fin, bool relu, float* Y) {
    k_rowsum_rsqrt<<<n, 128, 0, stream>>>(A, D, n);
    long nx = (long)n * fin;
    k_f32_to_f16<<<CDIV(nx, 256), 256, 0, stream>>>(X, XH, nx);
    k_transpose_w<<<CDIV(fin * HD, 256), 256, 0, stream>>>(W, WHT, fin);
    gemm(XH, WHT, XW, n, HD, fin);
    long nh = (long)n * HD;
    k_scale_rows<<<CDIV(nh, 256), 256, 0, stream>>>(XW, D, Z, ZHT, n);
    long nn = (long)n * n;
    k_f32_to_f16<<<CDIV(nn, 256), 256, 0, stream>>>(A, AH, nn);
    gemm(AH, ZHT, XW, n, HD, n);                  // AZ reuses XW buffer
    k_gcn_epilogue<<<CDIV(nh, 256), 256, 0, stream>>>(XW, Z, D, bias, Y, nh, relu ? 1 : 0);
  };

  // TopK pooling level: S = A+I (symmetric); A2 = (S@S)*(1-eye); top-k by
  // tanh score; HT <- gathered/scaled features, An <- A2[perm][:,perm]
  auto pool = [&](const float* A, const float* Hf, const float* pw, int n, int* P, float* An) {
    long nn = (long)n * n;
    k_add_eye_f16<<<CDIV(nn, 256), 256, 0, stream>>>(A, AH, n);
    gemm(AH, AH, ASQ, n, n, n);                   // S symmetric -> Bt = S
    k_zero_diag<<<CDIV(n, 256), 256, 0, stream>>>(ASQ, n);
    k_invnorm<<<1, 128, 0, stream>>>(pw, INV);
    k_scores<<<CDIV(n, 256), 256, 0, stream>>>(Hf, pw, INV, SC, n);
    k_bitonic<<<1, 1024, 0, stream>>>(SC, n, P);
    int k = n / 2;
    k_gather_pool<<<CDIV((long)k * HD, 256), 256, 0, stream>>>(Hf, SC, P, HT, k);
    k_gather_adj<<<CDIV((long)k * k, 256), 256, 0, stream>>>(ASQ, P, An, n, k);
  };

  // ---- build dense adjacency ----
  k_fill_f32<<<CDIV(4096L * 4096, 256), 256, 0, stream>>>(A0, 4096L * 4096, 0.0f);
  k_build_adj<<<CDIV(ne, 256), 256, 0, stream>>>(ei, A0, 4096, ne);

  // ---- down path ----
  gcn(A0, x, conv0_w, conv0_b, 4096, 64, true, XS0);
  pool(A0, XS0, pool_w + 0 * HD, 4096, P0, A1);
  gcn(A1, HT, down_w + 0 * HD * HD, down_b + 0 * HD, 2048, HD, true, XS1);
  pool(A1, XS1, pool_w + 1 * HD, 2048, P1, A2);
  gcn(A2, HT, down_w + 1 * HD * HD, down_b + 1 * HD, 1024, HD, true, XS2);
  pool(A2, XS2, pool_w + 2 * HD, 1024, P2, A3);
  gcn(A3, HT, down_w + 2 * HD * HD, down_b + 2 * HD, 512, HD, true, HC);

  // ---- up path (sum skip connections) ----
  hipMemcpyAsync(HT, XS2, 1024UL * HD * 4, hipMemcpyDeviceToDevice, stream);
  k_scatter_add<<<CDIV(512L * HD, 256), 256, 0, stream>>>(HT, HC, P2, 512);
  gcn(A2, HT, up_w + 0 * HD * HD, up_b + 0 * HD, 1024, HD, true, HC);

  hipMemcpyAsync(HT, XS1, 2048UL * HD * 4, hipMemcpyDeviceToDevice, stream);
  k_scatter_add<<<CDIV(1024L * HD, 256), 256, 0, stream>>>(HT, HC, P1, 1024);
  gcn(A1, HT, up_w + 1 * HD * HD, up_b + 1 * HD, 2048, HD, true, HC);

  hipMemcpyAsync(HT, XS0, 4096UL * HD * 4, hipMemcpyDeviceToDevice, stream);
  k_scatter_add<<<CDIV(2048L * HD, 256), 256, 0, stream>>>(HT, HC, P0, 2048);
  gcn(A0, HT, up_w + 2 * HD * HD, up_b + 2 * HD, 4096, HD, false, HC);

  // ---- pooled readout + head ----
  k_fill_f32<<<CDIV((long)NG * HD, 256), 256, 0, stream>>>(G, (long)NG * HD, 0.0f);
  k_segsum<<<CDIV(4096L * HD, 256), 256, 0, stream>>>(HC, batch, G, 4096);
  k_head<<<1, 128, 0, stream>>>(G, lins_w, lins_b, bn_g, bn_b, out_w, out_b, out);
}
